// PrunedModel_63239098467001
// MI455X (gfx1250) — compile-verified, matrix operand order checked by reading, not run on hardware
//
#include <hip/hip_runtime.h>
#include <hip/hip_bf16.h>

// ---------------------------------------------------------------------------
// Pruned MLP block for MI455X (gfx1250, wave32, WMMA, async global->LDS)
//   K1: LayerNorm + gather + residual copy
//   K2: gate/up GEMM (bf16 WMMA 16x16x32), double-buffered async LDS staging
//   K3: down GEMM (bf16 WMMA) + scatter-add into residual
// ---------------------------------------------------------------------------

#define HID   4096
#define PRN   2048
#define INTER 5004
#define IPAD  5008            // act row stride padded to 16B multiple
#define MTOT  16384           // B*S = 4*4096

typedef __attribute__((ext_vector_type(16))) __bf16 v16bf;
typedef __attribute__((ext_vector_type(8)))  float  v8f;

// ---- CDNA5 async global->LDS copies (ASYNCcnt-tracked, bypass VGPRs) ------
__device__ __forceinline__ unsigned lds_off(const void* p) {
    // LDS aperture: LDS address is the low 32 bits of the pointer
    return (unsigned)(unsigned long long)p;
}
__device__ __forceinline__ void async_ld_b128(unsigned lds, const void* g) {
    asm volatile("global_load_async_to_lds_b128 %0, %1, off"
                 :: "v"(lds), "v"(g) : "memory");
}
__device__ __forceinline__ void async_ld_b64(unsigned lds, const void* g) {
    asm volatile("global_load_async_to_lds_b64 %0, %1, off"
                 :: "v"(lds), "v"(g) : "memory");
}
__device__ __forceinline__ void wait_async_le4() {
#if __has_builtin(__builtin_amdgcn_s_wait_asynccnt)
    __builtin_amdgcn_s_wait_asynccnt(4);
#else
    asm volatile("s_wait_asynccnt 0x4" ::: "memory");
#endif
}
__device__ __forceinline__ void wait_async_0() {
#if __has_builtin(__builtin_amdgcn_s_wait_asynccnt)
    __builtin_amdgcn_s_wait_asynccnt(0);
#else
    asm volatile("s_wait_asynccnt 0x0" ::: "memory");
#endif
}

// Load a 16x32 bf16 fragment (A layout; [N][K] row-major B operands use the
// same per-lane pattern) from an LDS tile with row stride 32 elements.
// lane<16: row=lane,     K = {0..7, 16..23}
// lane>=16: row=lane-16, K = {8..15, 24..31}
__device__ __forceinline__ v16bf ld_frag_lds(const __hip_bfloat16* base) {
    const int lane = threadIdx.x & 31;
    const int r  = lane & 15;
    const int kb = (lane >> 4) << 3;           // 0 or 8
    const uint4* p = reinterpret_cast<const uint4*>(base + r * 32 + kb);
    union { v16bf v; uint4 q[2]; } f;
    f.q[0] = p[0];                             // K = kb .. kb+7
    f.q[1] = p[2];                             // K = kb+16 .. kb+23 (+32B)
    return f.v;
}

__device__ __forceinline__ v8f wmma_bf16(v16bf a, v16bf b, v8f c) {
    return __builtin_amdgcn_wmma_f32_16x16x32_bf16(
        /*neg_a=*/false, a, /*neg_b=*/false, b,
        /*c_mod=*/(short)0, c, /*reuse_a=*/false, /*reuse_b=*/false);
}

// ---------------------------------------------------------------------------
// K1: per-row LayerNorm (fp32 stats) + gather(s3) -> xg, residual copy -> out
// grid = MTOT blocks, 256 threads
// ---------------------------------------------------------------------------
__global__ void __launch_bounds__(256)
ln_gather_kernel(const __hip_bfloat16* __restrict__ hs,
                 const __hip_bfloat16* __restrict__ lnw,
                 const __hip_bfloat16* __restrict__ lnb,
                 const int* __restrict__ s3,
                 __hip_bfloat16* __restrict__ xg,
                 __hip_bfloat16* __restrict__ out) {
    __shared__ __align__(16) __hip_bfloat16 srow[HID];   // 8 KB raw row
    __shared__ float red1[256];
    __shared__ float red2[256];
    __shared__ float s_mu, s_inv;

    const int m   = blockIdx.x;
    const int tid = threadIdx.x;

    const __hip_bfloat16* row  = hs  + (size_t)m * HID;
    __hip_bfloat16*       orow = out + (size_t)m * HID;

    const uint4* rv = reinterpret_cast<const uint4*>(row);
    uint4*       ov = reinterpret_cast<uint4*>(orow);
    uint4*       sv = reinterpret_cast<uint4*>(srow);

    float sum = 0.f, sq = 0.f;
#pragma unroll
    for (int i = 0; i < 2; ++i) {
        const int idx = tid * 2 + i;           // 512 x uint4 = 4096 bf16
        uint4 q = rv[idx];
        ov[idx] = q;                           // residual copy
        sv[idx] = q;                           // stage for gather
        const __hip_bfloat16* e = reinterpret_cast<const __hip_bfloat16*>(&q);
#pragma unroll
        for (int j = 0; j < 8; ++j) {
            float f = __bfloat162float(e[j]);
            sum += f;
            sq  += f * f;
        }
    }

    red1[tid] = sum;
    red2[tid] = sq;
    __syncthreads();
#pragma unroll
    for (int s = 128; s > 0; s >>= 1) {
        if (tid < s) {
            red1[tid] += red1[tid + s];
            red2[tid] += red2[tid + s];
        }
        __syncthreads();
    }
    if (tid == 0) {
        float mu  = red1[0] * (1.0f / HID);
        float var = red2[0] * (1.0f / HID) - mu * mu;
        s_mu  = mu;
        s_inv = rsqrtf(var + 1e-5f);
    }
    __syncthreads();

    const float mu = s_mu, inv = s_inv;
    for (int p = tid; p < PRN; p += 256) {
        int idx = s3[p];
        float f = (__bfloat162float(srow[idx]) - mu) * inv;
        f = f * __bfloat162float(lnw[idx]) + __bfloat162float(lnb[idx]);
        xg[(size_t)m * PRN + p] = __float2bfloat16(f);
    }
}

// ---------------------------------------------------------------------------
// K2: act = sigmoid(xg @ gate_w^T + gb) * relu(xg @ up_w^T + ub)
// block tile 128(M) x 64(N), 8 waves; double-buffered async LDS staging.
// grid = (ceil(INTER/64)=79, MTOT/128=128)
// ---------------------------------------------------------------------------
__global__ void __launch_bounds__(256)
gateup_kernel(const __hip_bfloat16* __restrict__ xg,
              const __hip_bfloat16* __restrict__ gw,
              const __hip_bfloat16* __restrict__ gb,
              const __hip_bfloat16* __restrict__ uw,
              const __hip_bfloat16* __restrict__ ub,
              __hip_bfloat16* __restrict__ act) {
    __shared__ __align__(16) __hip_bfloat16 As[2][128 * 32];  // 2 x 8 KB
    __shared__ __align__(16) __hip_bfloat16 Bg[2][64 * 32];   // 2 x 4 KB
    __shared__ __align__(16) __hip_bfloat16 Bu[2][64 * 32];   // 2 x 4 KB

    const int tid  = threadIdx.x;
    const int wave = tid >> 5;
    const int lane = tid & 31;
    const int m0   = blockIdx.y * 128;
    const int n0   = blockIdx.x * 64;

    v8f accg[4] = {v8f{}, v8f{}, v8f{}, v8f{}};
    v8f accu[4] = {v8f{}, v8f{}, v8f{}, v8f{}};

    const int ar = tid >> 1;                 // A stage: row, 2 thr/row
    const int ah = (tid & 1) * 16;           // 16 elems (2 x b128) each
    const int br = tid >> 2;                 // B stage: row, 4 thr/row
    const int bs = (tid & 3) * 8;            // 8 elems (1 x b128) each

    // Clamp OOB N rows instead of zero-filling: those LDS rows only feed
    // output columns n >= INTER, which the epilogue discards. This keeps
    // async issue wave-uniform (no divergence -> exact ASYNCcnt).
    int nrow = n0 + br;
    if (nrow > INTER - 1) nrow = INTER - 1;

    const __hip_bfloat16* pa = xg + (size_t)(m0 + ar) * PRN + ah;
    const __hip_bfloat16* pg = gw + (size_t)nrow * PRN + bs;
    const __hip_bfloat16* pu = uw + (size_t)nrow * PRN + bs;

    unsigned lA[2], lG[2], lU[2];
#pragma unroll
    for (int b = 0; b < 2; ++b) {
        lA[b] = lds_off(&As[b][ar * 32 + ah]);
        lG[b] = lds_off(&Bg[b][br * 32 + bs]);
        lU[b] = lds_off(&Bu[b][br * 32 + bs]);
    }

    auto issue = [&](int k, int b) {          // 4 async ops per thread
        async_ld_b128(lA[b],      pa + k);
        async_ld_b128(lA[b] + 16, pa + k + 8);
        async_ld_b128(lG[b],      pg + k);
        async_ld_b128(lU[b],      pu + k);
    };

    issue(0, 0);
    const int STEPS = PRN / 32;               // 64
    for (int i = 0; i < STEPS; ++i) {
        if (i + 1 < STEPS) {
            issue((i + 1) * 32, (i + 1) & 1); // prefetch next slice
            wait_async_le4();                 // current slice complete
        } else {
            wait_async_0();
        }
        __syncthreads();                      // all waves' staging visible

        const int b = i & 1;
        v16bf a  = ld_frag_lds(&As[b][wave * 16 * 32]);
        v16bf g0 = ld_frag_lds(&Bg[b][0 * 16 * 32]);
        v16bf g1 = ld_frag_lds(&Bg[b][1 * 16 * 32]);
        v16bf g2 = ld_frag_lds(&Bg[b][2 * 16 * 32]);
        v16bf g3 = ld_frag_lds(&Bg[b][3 * 16 * 32]);
        v16bf u0 = ld_frag_lds(&Bu[b][0 * 16 * 32]);
        v16bf u1 = ld_frag_lds(&Bu[b][1 * 16 * 32]);
        v16bf u2 = ld_frag_lds(&Bu[b][2 * 16 * 32]);
        v16bf u3 = ld_frag_lds(&Bu[b][3 * 16 * 32]);
        accg[0] = wmma_bf16(a, g0, accg[0]);
        accg[1] = wmma_bf16(a, g1, accg[1]);
        accg[2] = wmma_bf16(a, g2, accg[2]);
        accg[3] = wmma_bf16(a, g3, accg[3]);
        accu[0] = wmma_bf16(a, u0, accu[0]);
        accu[1] = wmma_bf16(a, u1, accu[1]);
        accu[2] = wmma_bf16(a, u2, accu[2]);
        accu[3] = wmma_bf16(a, u3, accu[3]);
        __syncthreads();                      // done reading buf b
    }

    // epilogue: sigmoid(g)*relu(u); zero-fill pad columns [INTER, IPAD)
    const int nlane = lane & 15;
    const int hi    = lane >> 4;
#pragma unroll
    for (int t = 0; t < 4; ++t) {
        const int n = n0 + t * 16 + nlane;
        if (n >= IPAD) continue;
        const bool valid = (n < INTER);
        const float gbv = valid ? __bfloat162float(gb[n]) : 0.f;
        const float ubv = valid ? __bfloat162float(ub[n]) : 0.f;
#pragma unroll
        for (int v = 0; v < 8; ++v) {
            const int m = m0 + wave * 16 + hi * 8 + v;
            float o = 0.f;
            if (valid) {
                float g = accg[t][v] + gbv;
                float u = accu[t][v] + ubv;
                float s = 1.f / (1.f + __expf(-g));
                o = s * fmaxf(u, 0.f);
            }
            act[(size_t)m * IPAD + n] = __float2bfloat16(o);
        }
    }
}

// ---------------------------------------------------------------------------
// K3: out[m, s5[p]] += act @ down_w^T + down_b
// block tile 128(M) x 64(P); 156 async K-steps + 1 guarded sync tail step.
// grid = (PRN/64=32, MTOT/128=128)
// ---------------------------------------------------------------------------
__global__ void __launch_bounds__(256)
down_kernel(const __hip_bfloat16* __restrict__ act,
            const __hip_bfloat16* __restrict__ dw,
            const __hip_bfloat16* __restrict__ db,
            const int* __restrict__ s5,
            __hip_bfloat16* __restrict__ out) {
    __shared__ __align__(16) __hip_bfloat16 As[2][128 * 32];  // 2 x 8 KB
    __shared__ __align__(16) __hip_bfloat16 Bd[2][64 * 32];   // 2 x 4 KB

    const int tid  = threadIdx.x;
    const int wave = tid >> 5;
    const int lane = tid & 31;
    const int m0   = blockIdx.y * 128;
    const int n0   = blockIdx.x * 64;

    v8f accd[4] = {v8f{}, v8f{}, v8f{}, v8f{}};

    const int ar = tid >> 1;
    const int ah = (tid & 1) * 16;
    const int br = tid >> 2;
    const int bs = (tid & 3) * 8;
    const uint4 z4 = {0u, 0u, 0u, 0u};
    const uint2 z2 = {0u, 0u};

    const __hip_bfloat16* pa = act + (size_t)(m0 + ar) * IPAD + ah;
    const __hip_bfloat16* pd = dw + (size_t)(n0 + br) * INTER + bs; // 8B-aligned

    unsigned lA[2], lD[2];
#pragma unroll
    for (int b = 0; b < 2; ++b) {
        lA[b] = lds_off(&As[b][ar * 32 + ah]);
        lD[b] = lds_off(&Bd[b][br * 32 + bs]);
    }

    auto issue = [&](int k, int b) {          // 4 async ops per thread
        async_ld_b128(lA[b],      pa + k);
        async_ld_b128(lA[b] + 16, pa + k + 8);
        async_ld_b64(lD[b],       pd + k);    // down_w rows only 8B aligned
        async_ld_b64(lD[b] + 8,   pd + k + 4);
    };

    auto compute = [&](int b) {
        v16bf a  = ld_frag_lds(&As[b][wave * 16 * 32]);
        v16bf d0 = ld_frag_lds(&Bd[b][0 * 16 * 32]);
        v16bf d1 = ld_frag_lds(&Bd[b][1 * 16 * 32]);
        v16bf d2 = ld_frag_lds(&Bd[b][2 * 16 * 32]);
        v16bf d3 = ld_frag_lds(&Bd[b][3 * 16 * 32]);
        accd[0] = wmma_bf16(a, d0, accd[0]);
        accd[1] = wmma_bf16(a, d1, accd[1]);
        accd[2] = wmma_bf16(a, d2, accd[2]);
        accd[3] = wmma_bf16(a, d3, accd[3]);
    };

    const int FULL = INTER / 32;              // 156 full K-steps
    issue(0, 0);
    for (int i = 0; i < FULL; ++i) {
        if (i + 1 < FULL) {
            issue((i + 1) * 32, (i + 1) & 1);
            wait_async_le4();
        } else {
            wait_async_0();
        }
        __syncthreads();
        compute(i & 1);
        __syncthreads();
    }

    // Tail step k0 = 4992 (cols 4992..5023): synchronous guarded staging.
    // act pad cols [5004,5008) are zeros (written by K2); A chunk >= IPAD and
    // down_w chunks >= INTER are zeroed here, so garbage never contributes.
    {
        const int k0 = FULL * 32;             // 4992
        const int c0 = k0 + ah;               // multiple of 16
        uint4* la = reinterpret_cast<uint4*>(&As[0][ar * 32 + ah]);
        if (c0 < IPAD) {
            const uint4* ga = reinterpret_cast<const uint4*>(pa + k0);
            la[0] = ga[0];
            la[1] = ga[1];
        } else {
            la[0] = z4;
            la[1] = z4;
        }
        const int c = k0 + bs;                // multiple of 8; INTER % 4 == 0
        uint2* lb = reinterpret_cast<uint2*>(&Bd[0][br * 32 + bs]);
        lb[0] = (c + 4 <= INTER)
                    ? *reinterpret_cast<const uint2*>(pd + k0) : z2;
        lb[1] = (c + 8 <= INTER)
                    ? *reinterpret_cast<const uint2*>(pd + k0 + 4) : z2;
        __syncthreads();
        compute(0);
    }

    // epilogue: scatter-add into residual (each (m,p) owned by one lane)
    const int nlane = lane & 15;
    const int hi    = lane >> 4;
#pragma unroll
    for (int t = 0; t < 4; ++t) {
        const int p   = n0 + t * 16 + nlane;  // always < PRN
        const float dbv = __bfloat162float(db[p]);
        const int col = s5[p];
#pragma unroll
        for (int v = 0; v < 8; ++v) {
            const int m = m0 + wave * 16 + hi * 8 + v;
            const size_t o = (size_t)m * HID + col;
            float val = accd[t][v] + dbv + __bfloat162float(out[o]);
            out[o] = __float2bfloat16(val);
        }
    }
}

// ---------------------------------------------------------------------------
extern "C" void kernel_launch(void* const* d_in, const int* in_sizes, int n_in,
                              void* d_out, int out_size, void* d_ws, size_t ws_size,
                              hipStream_t stream) {
    const __hip_bfloat16* hs  = (const __hip_bfloat16*)d_in[0];
    const __hip_bfloat16* lnw = (const __hip_bfloat16*)d_in[1];
    const __hip_bfloat16* lnb = (const __hip_bfloat16*)d_in[2];
    const __hip_bfloat16* gw  = (const __hip_bfloat16*)d_in[3];
    const __hip_bfloat16* gb  = (const __hip_bfloat16*)d_in[4];
    const __hip_bfloat16* uw  = (const __hip_bfloat16*)d_in[5];
    const __hip_bfloat16* ub  = (const __hip_bfloat16*)d_in[6];
    const __hip_bfloat16* dw  = (const __hip_bfloat16*)d_in[7];
    const __hip_bfloat16* db  = (const __hip_bfloat16*)d_in[8];
    const int*            s3  = (const int*)d_in[9];
    const int*            s5  = (const int*)d_in[10];
    __hip_bfloat16*       out = (__hip_bfloat16*)d_out;

    // workspace: xg [MTOT, PRN] bf16 (64 MB) + act [MTOT, IPAD] bf16 (157 MB)
    __hip_bfloat16* xg  = (__hip_bfloat16*)d_ws;
    __hip_bfloat16* act = xg + (size_t)MTOT * PRN;

    ln_gather_kernel<<<MTOT, 256, 0, stream>>>(hs, lnw, lnb, s3, xg, out);

    dim3 g2((INTER + 63) / 64, MTOT / 128);
    gateup_kernel<<<g2, 256, 0, stream>>>(xg, gw, gb, uw, ub, act);

    dim3 g3(PRN / 64, MTOT / 128);
    down_kernel<<<g3, 256, 0, stream>>>(act, dw, db, s5, out);
}
